// RP_W_70832600646016
// MI455X (gfx1250) — matrix operand order repaired and checked by reading, not compiled, not run on hardware
//
#include <hip/hip_runtime.h>

// ---------------------------------------------------------------------------
// Types for CDNA5 WMMA (wave32, 16x16x32 bf16 -> f32)
// ---------------------------------------------------------------------------
typedef __bf16 bf16_t;
typedef __bf16 v16bf __attribute__((ext_vector_type(16)));
typedef __bf16 v8bf  __attribute__((ext_vector_type(8)));
typedef __bf16 v2bf  __attribute__((ext_vector_type(2)));
typedef float  v8f   __attribute__((ext_vector_type(8)));

// Half-band Kaiser(beta=2.5)-sinc taps (fc=0.5pi, 17 taps, even taps == 0)
#define FLT_C1  0.31357765f
#define FLT_C3 -0.09245783f
#define FLT_C5  0.04265465f
#define FLT_C7 -0.01936686f
#define SQRT2   1.41421356237f
#define LRELU_S 0.1f

// Geometry
#define TN        128          // final output positions per block
#define XS_L      162          // x positions staged: l in [n0-9, n0+152]
#define XS_STRIDE 72           // bf16 elems (144B rows; conflict-friendly)
#define W_STRIDE  72
#define Y_STRIDE  165          // f32 elems, gcd(165,64)=1
#define AO_STRIDE 137          // holds 135 odd-phase values
#define XS_BYTES  (XS_L * XS_STRIDE * 2)        // 23328
#define W_BYTES   (3 * 64 * W_STRIDE * 2)       // 27648
#define Y_BYTES   (64 * Y_STRIDE * 4)           // 42240
#define AO_BYTES  (64 * AO_STRIDE * 4)          // 35072
#define SMEM_BYTES (XS_BYTES + W_BYTES + Y_BYTES + AO_BYTES)  // 128288

// ---------------------------------------------------------------------------
// Prep 1: cond[b,c] = w[b,:] . affine_w[c,:] + affine_b[c]
// ---------------------------------------------------------------------------
__global__ __launch_bounds__(512) void cond_kernel(
    const float* __restrict__ w, const float* __restrict__ aw,
    const float* __restrict__ ab, float* __restrict__ cond) {
  __shared__ float wrow[512];
  const int b = blockIdx.x, c = threadIdx.x;
  wrow[c] = w[b * 512 + c];
  __syncthreads();
  const float* a = aw + (size_t)c * 512;
  float acc = 0.f;
  #pragma unroll 8
  for (int j = 0; j < 512; ++j) acc = fmaf(wrow[j], a[j], acc);
  cond[b * 512 + c] = acc + ab[c];
}

// ---------------------------------------------------------------------------
// Prep 2: sscale = rsqrt(mean(cond^2))   (deterministic tree reduction)
// ---------------------------------------------------------------------------
__global__ __launch_bounds__(256) void sscale_kernel(
    const float* __restrict__ cond, float* __restrict__ sscale) {
  __shared__ float red[256];
  const int t = threadIdx.x;
  float s = 0.f;
  for (int i = t; i < 8192; i += 256) { float v = cond[i]; s = fmaf(v, v, s); }
  red[t] = s;
  __syncthreads();
  for (int off = 128; off > 0; off >>= 1) {
    if (t < off) red[t] += red[t + off];
    __syncthreads();
  }
  if (t == 0) sscale[0] = rsqrtf(red[0] * (1.0f / 8192.0f));
}

// ---------------------------------------------------------------------------
// Prep 3: per-o weight RMS norm; pack Wt[k][o][i] (bf16); Wsq[o][i]=sum_k Wn^2
// conv_w layout: [o][i][k], k=3
// ---------------------------------------------------------------------------
__global__ __launch_bounds__(256) void wprep_kernel(
    const float* __restrict__ cw, bf16_t* __restrict__ wt,
    float* __restrict__ wsq) {
  __shared__ float red[256];
  const int o = blockIdx.x, t = threadIdx.x;
  const float* row = cw + (size_t)o * 1536;
  float s = 0.f;
  for (int i = t; i < 1536; i += 256) { float v = row[i]; s = fmaf(v, v, s); }
  red[t] = s;
  __syncthreads();
  for (int off = 128; off > 0; off >>= 1) {
    if (t < off) red[t] += red[t + off];
    __syncthreads();
  }
  const float f = rsqrtf(red[0] * (1.0f / 1536.0f));
  for (int i = t; i < 512; i += 256) {
    float w0 = row[i * 3 + 0] * f;
    float w1 = row[i * 3 + 1] * f;
    float w2 = row[i * 3 + 2] * f;
    wt[((size_t)0 * 512 + o) * 512 + i] = (bf16_t)w0;
    wt[((size_t)1 * 512 + o) * 512 + i] = (bf16_t)w1;
    wt[((size_t)2 * 512 + o) * 512 + i] = (bf16_t)w2;
    wsq[o * 512 + i] = w0 * w0 + w1 * w1 + w2 * w2;
  }
}

// ---------------------------------------------------------------------------
// Prep 4: d[b,o] = rsqrt(sum_i Wsq[o,i] * (cond[b,i]*sscale)^2 + 1e-8)
// ---------------------------------------------------------------------------
__global__ __launch_bounds__(512) void demod_kernel(
    const float* __restrict__ cond, const float* __restrict__ sscale,
    const float* __restrict__ wsq, float* __restrict__ dcoef) {
  __shared__ float s2[512];
  const int b = blockIdx.x, t = threadIdx.x;
  const float ss = sscale[0];
  float c = cond[b * 512 + t] * ss;
  s2[t] = c * c;
  __syncthreads();
  const float* wr = wsq + (size_t)t * 512;
  float acc = 1e-8f;
  #pragma unroll 8
  for (int i = 0; i < 512; ++i) acc = fmaf(wr[i], s2[i], acc);
  dcoef[b * 512 + t] = rsqrtf(acc);
}

// ---------------------------------------------------------------------------
// Main fused kernel: modulated conv1d (bf16 WMMA, f32 acc) + demod + bias
//   + zero-insert x2 upsample + halfband LPF + lrelu*sqrt2 + LPF + /2 decimate
// Grid: (L/128, C_out/64, B). 256 threads = 8 wave32.
// ---------------------------------------------------------------------------
__global__ __launch_bounds__(256) void modconv_resample_kernel(
    const float* __restrict__ x, const float* __restrict__ cond,
    const float* __restrict__ sscale, const float* __restrict__ dcoef,
    const float* __restrict__ conv_b, const float* __restrict__ norm_ema,
    const bf16_t* __restrict__ wt, float* __restrict__ out) {
  extern __shared__ __align__(16) char smem[];
  bf16_t* lds_xs = (bf16_t*)smem;                              // [XS_L][XS_STRIDE]
  bf16_t* lds_w  = (bf16_t*)(smem + XS_BYTES);                 // [3][64][W_STRIDE]
  float*  lds_y  = (float*)(smem + XS_BYTES + W_BYTES);        // [64][Y_STRIDE]
  float*  lds_ao = (float*)(smem + XS_BYTES + W_BYTES + Y_BYTES); // [64][AO_STRIDE]

  const int t    = threadIdx.x;
  const int lane = t & 31;
  const int wave = t >> 5;
  const int n0 = blockIdx.x * TN;   // output tile start (y tile starts at n0-8)
  const int o0 = blockIdx.y * 64;
  const int b  = blockIdx.z;

  const float smul = sscale[0] * rsqrtf(norm_ema[0]);  // cond -> conv-input scale

  // Wave tiling: 4 M-tiles (16 rows of o) x 2 N-halves (5 tiles of 16 l each)
  const int m_tile = wave >> 1;
  const int nhalf  = wave & 1;
  const int ln15   = lane & 15;
  const int lhi    = lane >> 4;

  v8f acc[5];
  #pragma unroll
  for (int nt = 0; nt < 5; ++nt) acc[nt] = (v8f)(0.f);

  // ------------------- K loop over 8 chunks of 64 input channels -----------
  for (int ch = 0; ch < 8; ++ch) {
    const int i0 = ch * 64;

    // Stage xs[l][i] = x[b, i, n0-9+l] * cond[b,i]*smul  (bf16, transposed).
    // Load-all / store-all phases keep up to 12 global loads in flight.
    #pragma unroll
    for (int pass = 0; pass < 4; ++pass) {
      const int ii = (wave + pass * 8) * 2;           // 0..62, even (wave-uniform)
      const int gi = i0 + ii;
      const float sv0 = cond[b * 512 + gi] * smul;
      const float sv1 = cond[b * 512 + gi + 1] * smul;
      const float* xr0 = x + ((size_t)b * 512 + gi) * 2048;
      const float* xr1 = xr0 + 2048;
      float v0[6], v1[6];
      #pragma unroll
      for (int r = 0; r < 6; ++r) {
        const int l = lane + r * 32;
        const int p = n0 - 9 + l;
        const bool in = (l < XS_L) && (p >= 0) && (p < 2048);
        v0[r] = in ? xr0[p] : 0.f;
        v1[r] = in ? xr1[p] : 0.f;
      }
      #pragma unroll
      for (int r = 0; r < 6; ++r) {
        const int l = lane + r * 32;
        if (l < XS_L) {
          v2bf pk;
          pk[0] = (bf16_t)(v0[r] * sv0);
          pk[1] = (bf16_t)(v1[r] * sv1);
          *(v2bf*)&lds_xs[l * XS_STRIDE + ii] = pk;
        }
      }
    }
    // Stage weights: lds_w[k][oo][i] from wt[k][o0+oo][i0+i].
    // 6 x 16B loads in flight, then 6 stores (1536 rows == 6*256 exactly).
    {
      v8bf wbuf[6];
      #pragma unroll
      for (int r = 0; r < 6; ++r) {
        const int dd = t + r * 256;
        const int k  = dd / 512;
        const int rm = dd - k * 512;
        const int oo = rm >> 3;
        const int i8 = (rm & 7) << 3;
        wbuf[r] = *(const v8bf*)(wt + ((size_t)(k * 512 + o0 + oo)) * 512 + i0 + i8);
      }
      #pragma unroll
      for (int r = 0; r < 6; ++r) {
        const int dd = t + r * 256;
        const int k  = dd / 512;
        const int rm = dd - k * 512;
        const int oo = rm >> 3;
        const int i8 = (rm & 7) << 3;
        *(v8bf*)&lds_w[(k * 64 + oo) * W_STRIDE + i8] = wbuf[r];
      }
    }
    __syncthreads();

    // GEMM on this chunk: K = 64 channels (2 wmma steps) x 3 taps.
    // sched_barrier(0) pins the 12 ds_loads of a group above its 5 WMMAs
    // (distinct dest registers, one dscnt wait), while the next group's
    // loads remain free to interleave with these WMMAs.
    const int a_koff = lhi * 8;    // A: per-lane K runs base, base+16
  const int b_koff = lhi * 8;    // B: per-lane K runs base, base+16
    const int aml = m_tile * 16 + ln15;
    #pragma unroll
    for (int ci = 0; ci < 2; ++ci) {
      #pragma unroll
      for (int k = 0; k < 3; ++k) {
        const bf16_t* ap = &lds_w[(k * 64 + aml) * W_STRIDE + ci * 32 + a_koff];
        const v8bf a_lo = *(const v8bf*)ap;
        const v8bf a_hi = *(const v8bf*)(ap + 16);
        const v16bf A = __builtin_shufflevector(a_lo, a_hi,
            0, 1, 2, 3, 4, 5, 6, 7, 8, 9, 10, 11, 12, 13, 14, 15);
        v16bf Bv[5];
        #pragma unroll
        for (int nt = 0; nt < 5; ++nt) {
          const int lrel = nhalf * 80 + nt * 16 + ln15;
          const bf16_t* bp = &lds_xs[(lrel + k) * XS_STRIDE + ci * 32 + b_koff];
          const v8bf b_lo = *(const v8bf*)bp;
          const v8bf b_hi = *(const v8bf*)(bp + 16);
          Bv[nt] = __builtin_shufflevector(b_lo, b_hi,
              0, 1, 2, 3, 4, 5, 6, 7, 8, 9, 10, 11, 12, 13, 14, 15);
        }
        __builtin_amdgcn_sched_barrier(0);   // loads above, WMMAs below
        #pragma unroll
        for (int nt = 0; nt < 5; ++nt) {
          acc[nt] = __builtin_amdgcn_wmma_f32_16x16x32_bf16(
              false, A, false, Bv[nt], (short)0, acc[nt], false, false);
        }
      }
    }
    __syncthreads();
  }

  // ------------------- Epilogue: y = acc*d + bias -> LDS --------------------
  float dm[8], bm[8];
  #pragma unroll
  for (int r = 0; r < 8; ++r) {
    const int o_loc = m_tile * 16 + lhi * 8 + r;   // C/D layout: M = r + 8*(lane>=16)
    dm[r] = dcoef[b * 512 + o0 + o_loc];
    bm[r] = conv_b[o0 + o_loc];
  }
  #pragma unroll
  for (int nt = 0; nt < 5; ++nt) {
    const int lrel = nhalf * 80 + nt * 16 + ln15;
    const int j_abs = n0 - 8 + lrel;
    const bool valid = (j_abs >= 0) && (j_abs < 2048);  // zero-pad semantics
    #pragma unroll
    for (int r = 0; r < 8; ++r) {
      const int o_loc = m_tile * 16 + lhi * 8 + r;
      const float v = valid ? fmaf(acc[nt][r], dm[r], bm[r]) : 0.f;
      lds_y[o_loc * Y_STRIDE + lrel] = v;
    }
  }
  __syncthreads();

  // ---- Phase A: odd-phase upsample + lrelu:  ao[j] over j_abs in [n0-4, n0+130]
  for (int idx = t; idx < 64 * 135; idx += 256) {
    const int oo   = idx / 135;
    const int jj   = idx - oo * 135;       // 0..134
    const int jrel = jj + 4;               // y index (rel to n0-8)
    const int j_abs = n0 - 8 + jrel;
    const float* yr = &lds_y[oo * Y_STRIDE + jrel];
    float h = FLT_C1 * (yr[0] + yr[1]);
    h = fmaf(FLT_C3, yr[-1] + yr[2], h);
    h = fmaf(FLT_C5, yr[-2] + yr[3], h);
    h = fmaf(FLT_C7, yr[-3] + yr[4], h);
    const float a = (h > 0.f ? h : LRELU_S * h) * SQRT2;
    lds_ao[oo * AO_STRIDE + jj] = (j_abs >= 0 && j_abs < 2048) ? a : 0.f;
  }
  __syncthreads();

  // ---- Phase B: down-filter + decimate; out[b, o, n0+nr] ----
  for (int idx = t; idx < 64 * 128; idx += 256) {
    const int oo = idx >> 7;
    const int nr = idx & 127;
    const float yc = lds_y[oo * Y_STRIDE + nr + 8];
    // even-phase sample: a[2n] = lrelu(0.5*y[n])*sqrt2
    const float e = (yc > 0.f ? yc : LRELU_S * yc) * (0.5f * SQRT2);
    const float* ar = &lds_ao[oo * AO_STRIDE + nr];
    float v = 0.5f * e;
    v = fmaf(FLT_C1, ar[3] + ar[4], v);
    v = fmaf(FLT_C3, ar[2] + ar[5], v);
    v = fmaf(FLT_C5, ar[1] + ar[6], v);
    v = fmaf(FLT_C7, ar[0] + ar[7], v);
    out[((size_t)b * 512 + o0 + oo) * 2048 + n0 + nr] = v;
  }
}

// ---------------------------------------------------------------------------
extern "C" void kernel_launch(void* const* d_in, const int* in_sizes, int n_in,
                              void* d_out, int out_size, void* d_ws, size_t ws_size,
                              hipStream_t stream) {
  (void)in_sizes; (void)n_in; (void)out_size; (void)ws_size;
  const float* x   = (const float*)d_in[0];   // (16, 512, 2048)
  const float* w   = (const float*)d_in[1];   // (16, 512)
  const float* aw  = (const float*)d_in[2];   // (512, 512)
  const float* ab  = (const float*)d_in[3];   // (512,)
  const float* cw  = (const float*)d_in[4];   // (512, 512, 3)
  const float* cb  = (const float*)d_in[5];   // (512,)
  const float* nem = (const float*)d_in[6];   // scalar
  float* outp = (float*)d_out;                // (16, 512, 2048)

  // Workspace carve (~2.7 MB)
  float* cond   = (float*)d_ws;          // 8192
  float* sscale = cond + 8192;           // 1 (padded to 8)
  float* dcoef  = sscale + 8;            // 8192
  float* wsq    = dcoef + 8192;          // 262144
  bf16_t* wt    = (bf16_t*)(wsq + 262144); // 786432 bf16 packed [k][o][i]

  cond_kernel<<<16, 512, 0, stream>>>(w, aw, ab, cond);
  sscale_kernel<<<1, 256, 0, stream>>>(cond, sscale);
  wprep_kernel<<<512, 256, 0, stream>>>(cw, wt, wsq);
  demod_kernel<<<16, 512, 0, stream>>>(cond, sscale, wsq, dcoef);

  dim3 grid(2048 / TN, 512 / 64, 16);
  modconv_resample_kernel<<<grid, 256, SMEM_BYTES, stream>>>(
      x, cond, sscale, dcoef, cb, nem, wt, outp);
}
